// AttnBlock_13400297963922
// MI455X (gfx1250) — matrix operand order repaired and checked by reading, not compiled, not run on hardware
//
#include <hip/hip_runtime.h>
#include <hip/hip_bf16.h>

// ---------------------------------------------------------------------------
// Transformer block (pre-norm attn + pre-norm GELU FFN) for MI455X / gfx1250.
// Compute-bound (~258 GFLOP vs ~100MB traffic) => bf16 WMMA with f32 accum.
// Data movement: async global->LDS (ASYNCcnt) with double-buffered tiles.
// ---------------------------------------------------------------------------

constexpr int kB  = 2;
constexpr int kC  = 1024;
constexpr int kT  = 4096;
constexpr int kH  = 4;
constexpr int kDH = 256;     // kC / kH
constexpr int kFF = 1536;
constexpr int kM  = kB * kT; // 8192 rows

typedef __bf16 bf16_t;
typedef float  vf8   __attribute__((ext_vector_type(8)));
typedef __bf16 vbf16 __attribute__((ext_vector_type(16)));
typedef __bf16 vbf8  __attribute__((ext_vector_type(8)));

static __device__ inline vf8 wmma_bf16(vbf16 a, vbf16 b, vf8 c) {
  // D = A(16x32 bf16) * B(32x16 bf16) + C(16x16 f32)
  return __builtin_amdgcn_wmma_f32_16x16x32_bf16(
      /*neg_a=*/false, a, /*neg_b=*/false, b,
      /*c_mod=*/(short)0, c, /*reuse_a=*/false, /*reuse_b=*/false);
}

// Async DMA: 16B global -> LDS, bypasses VGPRs, tracked by ASYNCcnt.
// GV addressing mode (64-bit VGPR address), LDS byte address in VDST VGPR.
static __device__ inline void async_b128(uint32_t lds_off, const void* gaddr) {
  asm volatile("global_load_async_to_lds_b128 %0, %1, off"
               :: "v"(lds_off), "v"((uint64_t)(uintptr_t)gaddr)
               : "memory");
}
static __device__ inline void wait_async0() {
  asm volatile("s_wait_asynccnt 0" ::: "memory");
}
// generic (flat-aperture) shared pointer -> 32-bit LDS address (addr[31:0])
static __device__ inline uint32_t lds_addr(const void* p) {
  return (uint32_t)(uintptr_t)p;
}

// ---------------------------------------------------------------------------
// fp32 -> bf16 elementwise (weights)
// ---------------------------------------------------------------------------
__global__ void k_cvt(const float* __restrict__ s, bf16_t* __restrict__ d, int n) {
  int i = blockIdx.x * 256 + threadIdx.x;
  if (i < n) d[i] = (bf16_t)s[i];
}

// ---------------------------------------------------------------------------
// 32x32 tiled transpose: src[R][Cd] -> dst[Cd][R], batched over blockIdx.z
// ---------------------------------------------------------------------------
__global__ __launch_bounds__(256)
void k_transpose(const float* __restrict__ src, float* __restrict__ dst,
                 int R, int Cd) {
  __shared__ float tile[32][33];
  size_t base = (size_t)blockIdx.z * R * Cd;
  int r0 = blockIdx.y * 32, c0 = blockIdx.x * 32;
  int tx = threadIdx.x, ty = threadIdx.y; // 32 x 8
#pragma unroll
  for (int i = 0; i < 32; i += 8)
    tile[ty + i][tx] = src[base + (size_t)(r0 + ty + i) * Cd + c0 + tx];
  __syncthreads();
#pragma unroll
  for (int i = 0; i < 32; i += 8)
    dst[base + (size_t)(c0 + ty + i) * R + r0 + tx] = tile[tx][ty + i];
}

// ---------------------------------------------------------------------------
// RMSNorm over C=1024: one block (256 thr) per (b,t) row, bf16 output
// ---------------------------------------------------------------------------
__global__ __launch_bounds__(256)
void k_rmsnorm(const float* __restrict__ h, const float* __restrict__ g,
               bf16_t* __restrict__ out) {
  __shared__ float red[8];
  const size_t row = blockIdx.x;
  const float* hp = h + row * kC;
  float ss = 0.f;
#pragma unroll
  for (int c = threadIdx.x, i = 0; i < kC / 256; c += 256, ++i) {
    float v = hp[c];
    ss += v * v;
  }
#pragma unroll
  for (int off = 16; off > 0; off >>= 1) ss += __shfl_xor(ss, off, 32);
  if ((threadIdx.x & 31) == 0) red[threadIdx.x >> 5] = ss;
  __syncthreads();
  float tot = 0.f;
#pragma unroll
  for (int i = 0; i < 8; ++i) tot += red[i];
  float inv = rsqrtf(tot * (1.0f / kC) + 1e-8f);
#pragma unroll
  for (int c = threadIdx.x, i = 0; i < kC / 256; c += 256, ++i)
    out[row * kC + c] = (bf16_t)(hp[c] * inv * g[c]);
}

// ---------------------------------------------------------------------------
// Tiled WMMA GEMM: Out(M,N) = A(M,K) @ B(K,N), bf16 in, f32 accum.
// Block tile 128x128, BK=32, 256 threads = 8 waves (4x2), 8 accum tiles/wave.
// Double-buffered LDS; A tile via async-to-LDS DMA; B tile loaded to VGPRs and
// stored transposed [n][k] with packed b32 stores (k-pairs contiguous).
// EPI: 0 = bf16 store, 1 = transposed V store (B,H,DH,T), 2 = f32 +=, 3 = GELU->bf16
// ---------------------------------------------------------------------------
template <int EPI>
__global__ __launch_bounds__(256)
void k_gemm(const bf16_t* __restrict__ A, const bf16_t* __restrict__ Bw,
            void* __restrict__ OutP, int Ndim, int Kdim) {
  __shared__ alignas(16) bf16_t a_s[2][128 * 32];
  __shared__ alignas(16) bf16_t b_s[2][128 * 32]; // [n][k]

  const int tid = threadIdx.x;
  const int wave = tid >> 5, lane = tid & 31;
  const int half = lane >> 4, l16 = lane & 15;
  const int m0 = blockIdx.y * 128, n0 = blockIdx.x * 128;
  const int wm = (wave >> 1) * 32, wn = (wave & 1) * 64;

  vf8 acc[2][4];
#pragma unroll
  for (int i = 0; i < 2; ++i)
#pragma unroll
    for (int j = 0; j < 4; ++j)
#pragma unroll
      for (int e = 0; e < 8; ++e) acc[i][j][e] = 0.f;

  const int arow = tid >> 1, akk = (tid & 1) * 16; // A: 2 thr/row, 16B chunks
  const int bkp = tid >> 4, bnb = (tid & 15) * 8;  // B: k-pair per thread

  auto load_tile = [&](int k0, int buf) {
    // A slab 128x32: verbatim copy -> async DMA, 2 x 16B per thread
    const bf16_t* ag = &A[(size_t)(m0 + arow) * Kdim + k0 + akk];
    uint32_t la = lds_addr(&a_s[buf][arow * 32 + akk]);
    async_b128(la, ag);
    async_b128(la + 16, ag + 8);
    // B rows 2*bkp, 2*bkp+1 -> packed [n][k] scatter (8 x b32 stores)
    vbf8 c0 = *(const vbf8*)&Bw[(size_t)(k0 + 2 * bkp) * Ndim + n0 + bnb];
    vbf8 c1 = *(const vbf8*)&Bw[(size_t)(k0 + 2 * bkp + 1) * Ndim + n0 + bnb];
#pragma unroll
    for (int i = 0; i < 8; ++i) {
      uint32_t lo = (uint32_t)__builtin_bit_cast(uint16_t, c0[i]);
      uint32_t hi = (uint32_t)__builtin_bit_cast(uint16_t, c1[i]);
      *(uint32_t*)&b_s[buf][(bnb + i) * 32 + 2 * bkp] = lo | (hi << 16);
    }
  };

  const int nk = Kdim / 32;
  load_tile(0, 0);
  for (int it = 0; it < nk; ++it) {
    const int buf = it & 1;
    wait_async0();
    __syncthreads();
    if (it + 1 < nk) load_tile((it + 1) * 32, buf ^ 1);

    vbf16 af[2];
#pragma unroll
    for (int i = 0; i < 2; ++i) {
      const bf16_t* ap = &a_s[buf][(wm + i * 16 + l16) * 32];
      ((vbf8*)&af[i])[0] = *(const vbf8*)(ap + half * 16);       // K 0-7 / 16-23
      ((vbf8*)&af[i])[1] = *(const vbf8*)(ap + 8 + half * 16);  // K 8-15 / 24-31
    }
#pragma unroll
    for (int jt = 0; jt < 4; ++jt) {
      vbf16 bfr;
      const bf16_t* bp = &b_s[buf][(wn + jt * 16 + l16) * 32 + half * 16];
      ((vbf8*)&bfr)[0] = *(const vbf8*)bp;        // K 0-15 or 16-31, contiguous
      ((vbf8*)&bfr)[1] = *(const vbf8*)(bp + 8);
#pragma unroll
      for (int i = 0; i < 2; ++i) acc[i][jt] = wmma_bf16(af[i], bfr, acc[i][jt]);
    }
  }

#pragma unroll
  for (int i = 0; i < 2; ++i)
#pragma unroll
    for (int jt = 0; jt < 4; ++jt)
#pragma unroll
      for (int r = 0; r < 8; ++r) {
        int row = m0 + wm + i * 16 + r + half * 8; // C/D layout: VGPR r <-> M r/r+8
        int col = n0 + wn + jt * 16 + l16;
        float v = acc[i][jt][r];
        if (EPI == 0) {
          ((bf16_t*)OutP)[(size_t)row * Ndim + col] = (bf16_t)v;
        } else if (EPI == 1) { // V^T store: (B,H,DH,T)
          int bb = row >> 12, t = row & (kT - 1);
          int hd = col >> 8, dh = col & (kDH - 1);
          ((bf16_t*)OutP)[((size_t)(bb * kH + hd) * kDH + dh) * kT + t] = (bf16_t)v;
        } else if (EPI == 2) {
          ((float*)OutP)[(size_t)row * Ndim + col] += v;
        } else { // exact GELU
          float g = 0.5f * v * (1.0f + erff(v * 0.70710678118f));
          ((bf16_t*)OutP)[(size_t)row * Ndim + col] = (bf16_t)g;
        }
      }
}

// ---------------------------------------------------------------------------
// Flash attention, DH=256, 4 waves/block, 16 q-rows per wave, 32-key tiles.
// Q fragments resident in VGPRs; K/V^T tiles double-buffered in LDS via async
// DMA; online softmax with width-16 shuffles matching the C/D layout.
// ---------------------------------------------------------------------------
__global__ __launch_bounds__(128)
void k_flash(const bf16_t* __restrict__ Qb, const bf16_t* __restrict__ Kb,
             const bf16_t* __restrict__ VTb, bf16_t* __restrict__ Ob) {
  __shared__ alignas(16) bf16_t k_s[2][32 * kDH];  // [key][dh]   2x16KB
  __shared__ alignas(16) bf16_t vt_s[2][kDH * 32]; // [dh][key]   2x16KB
  __shared__ alignas(16) bf16_t p_s[4][16 * 32];   // per-wave P     4KB

  const int tid = threadIdx.x;
  const int wave = tid >> 5, lane = tid & 31;
  const int half = lane >> 4, l16 = lane & 15;
  const int bh = blockIdx.y, b = bh >> 2, col0 = (bh & 3) * kDH;
  const int q0 = blockIdx.x * 64;

  // Q fragments (16 rows x 256 dh) -> 8 x v16bf in registers
  vbf16 qf[8];
  const bf16_t* qbase = Qb + (size_t)(b * kT + q0 + wave * 16 + l16) * kC + col0;
#pragma unroll
  for (int j = 0; j < 8; ++j) {
    ((vbf8*)&qf[j])[0] = *(const vbf8*)(qbase + j * 32 + half * 16);
    ((vbf8*)&qf[j])[1] = *(const vbf8*)(qbase + j * 32 + 8 + half * 16);
  }

  vf8 o[16]; // O accum: 16 dh-tiles of 16x16
#pragma unroll
  for (int t = 0; t < 16; ++t)
#pragma unroll
    for (int e = 0; e < 8; ++e) o[t][e] = 0.f;
  float mrow[8], lrow[8];
#pragma unroll
  for (int r = 0; r < 8; ++r) { mrow[r] = -3.0e38f; lrow[r] = 0.f; }

  auto load_kv = [&](int k0, int buf) {
#pragma unroll
    for (int i = 0; i < 8; ++i) { // K tile 32x256 (verbatim rows)
      int ch = tid + 128 * i;
      int kr = ch >> 5, dh = (ch & 31) << 3;
      async_b128(lds_addr(&k_s[buf][kr * kDH + dh]),
                 &Kb[(size_t)(b * kT + k0 + kr) * kC + col0 + dh]);
    }
#pragma unroll
    for (int i = 0; i < 8; ++i) { // V^T tile 256x32 (verbatim rows)
      int ch = tid + 128 * i;
      int dh = ch >> 2, kk = (ch & 3) << 3;
      async_b128(lds_addr(&vt_s[buf][dh * 32 + kk]),
                 &VTb[(size_t)(bh * kDH + dh) * kT + k0 + kk]);
    }
  };

  load_kv(0, 0);
  for (int k0 = 0; k0 < kT; k0 += 32) {
    const int buf = (k0 >> 5) & 1;
    wait_async0();
    __syncthreads();
    if (k0 + 32 < kT) load_kv(k0 + 32, buf ^ 1);

    // S = Q @ K^T : two 16x16 key tiles, K-dim 256 = 8 WMMAs each
    vf8 s0, s1;
#pragma unroll
    for (int e = 0; e < 8; ++e) { s0[e] = 0.f; s1[e] = 0.f; }
#pragma unroll
    for (int j = 0; j < 8; ++j) {
      vbf16 kb0, kb1;
      const bf16_t* kp0 = &k_s[buf][l16 * kDH + j * 32 + half * 16];
      const bf16_t* kp1 = &k_s[buf][(16 + l16) * kDH + j * 32 + half * 16];
      ((vbf8*)&kb0)[0] = *(const vbf8*)kp0;
      ((vbf8*)&kb0)[1] = *(const vbf8*)(kp0 + 8);
      ((vbf8*)&kb1)[0] = *(const vbf8*)kp1;
      ((vbf8*)&kb1)[1] = *(const vbf8*)(kp1 + 8);
      s0 = wmma_bf16(qf[j], kb0, s0);
      s1 = wmma_bf16(qf[j], kb1, s1);
    }

    // online softmax (row r lives in VGPR r; lanes 0-15 = rows 0-7, 16-31 = 8-15)
#pragma unroll
    for (int r = 0; r < 8; ++r) {
      float x0 = s0[r] * 0.0625f, x1 = s1[r] * 0.0625f; // * DH^-0.5
      float mx = fmaxf(x0, x1);
#pragma unroll
      for (int off = 8; off > 0; off >>= 1) mx = fmaxf(mx, __shfl_xor(mx, off, 16));
      float mn = fmaxf(mrow[r], mx);
      float alpha = __expf(mrow[r] - mn);
      float p0 = __expf(x0 - mn), p1 = __expf(x1 - mn);
      float rs = p0 + p1;
#pragma unroll
      for (int off = 8; off > 0; off >>= 1) rs += __shfl_xor(rs, off, 16);
      lrow[r] = lrow[r] * alpha + rs;
      mrow[r] = mn;
#pragma unroll
      for (int t = 0; t < 16; ++t) o[t][r] *= alpha;
      p_s[wave][(r + half * 8) * 32 + l16] = (bf16_t)p0;
      p_s[wave][(r + half * 8) * 32 + 16 + l16] = (bf16_t)p1;
    }
    asm volatile("s_wait_dscnt 0" ::: "memory"); // same-wave LDS store->load

    // O += P(16x32) @ V(32x256)
    vbf16 pa;
    ((vbf8*)&pa)[0] = *(const vbf8*)&p_s[wave][l16 * 32 + half * 16];
    ((vbf8*)&pa)[1] = *(const vbf8*)&p_s[wave][l16 * 32 + 8 + half * 16];
#pragma unroll
    for (int t = 0; t < 16; ++t) {
      vbf16 vb;
      const bf16_t* vp = &vt_s[buf][(t * 16 + l16) * 32 + half * 16];
      ((vbf8*)&vb)[0] = *(const vbf8*)vp;
      ((vbf8*)&vb)[1] = *(const vbf8*)(vp + 8);
      o[t] = wmma_bf16(pa, vb, o[t]);
    }
  }

  // normalize and store (bf16, (B,T,C) layout for the Wo GEMM)
#pragma unroll
  for (int t = 0; t < 16; ++t)
#pragma unroll
    for (int r = 0; r < 8; ++r) {
      float val = o[t][r] / lrow[r];
      int row = q0 + wave * 16 + r + half * 8;
      Ob[(size_t)(b * kT + row) * kC + col0 + t * 16 + l16] = (bf16_t)val;
    }
}

// ---------------------------------------------------------------------------
extern "C" void kernel_launch(void* const* d_in, const int* in_sizes, int n_in,
                              void* d_out, int out_size, void* d_ws, size_t ws_size,
                              hipStream_t stream) {
  (void)in_sizes; (void)n_in; (void)out_size; (void)ws_size;
  const float* x      = (const float*)d_in[0];
  const float* g_attn = (const float*)d_in[1];
  const float* Wq     = (const float*)d_in[2];
  const float* Wk     = (const float*)d_in[3];
  const float* Wv     = (const float*)d_in[4];
  const float* Wo     = (const float*)d_in[5];
  const float* g_ff   = (const float*)d_in[6];
  const float* W1     = (const float*)d_in[7];
  const float* W2     = (const float*)d_in[8];
  float* out = (float*)d_out;

  char* ws = (char*)d_ws;
  size_t off = 0;
  float*  h    = (float*)(ws + off);  off += (size_t)kM * kC * 4;
  bf16_t* wq_b = (bf16_t*)(ws + off); off += (size_t)kC * kC * 2;
  bf16_t* wk_b = (bf16_t*)(ws + off); off += (size_t)kC * kC * 2;
  bf16_t* wv_b = (bf16_t*)(ws + off); off += (size_t)kC * kC * 2;
  bf16_t* wo_b = (bf16_t*)(ws + off); off += (size_t)kC * kC * 2;
  bf16_t* w1_b = (bf16_t*)(ws + off); off += (size_t)kC * kFF * 2;
  bf16_t* w2_b = (bf16_t*)(ws + off); off += (size_t)kFF * kC * 2;
  bf16_t* a_b  = (bf16_t*)(ws + off); off += (size_t)kM * kC * 2;
  bf16_t* q_b  = (bf16_t*)(ws + off); off += (size_t)kM * kC * 2;
  bf16_t* k_b  = (bf16_t*)(ws + off); off += (size_t)kM * kC * 2;
  bf16_t* vt_b = (bf16_t*)(ws + off); off += (size_t)kM * kC * 2;
  bf16_t* at_b = (bf16_t*)(ws + off); off += (size_t)kM * kC * 2;
  bf16_t* f_b  = a_b;  // reuse: a dead after QKV
  bf16_t* ff_b = q_b;  // reuse: q+k region (33.5MB) holds ff acts (25MB)

  // weights -> bf16
  k_cvt<<<dim3(kC * kC / 256), 256, 0, stream>>>(Wq, wq_b, kC * kC);
  k_cvt<<<dim3(kC * kC / 256), 256, 0, stream>>>(Wk, wk_b, kC * kC);
  k_cvt<<<dim3(kC * kC / 256), 256, 0, stream>>>(Wv, wv_b, kC * kC);
  k_cvt<<<dim3(kC * kC / 256), 256, 0, stream>>>(Wo, wo_b, kC * kC);
  k_cvt<<<dim3(kC * kFF / 256), 256, 0, stream>>>(W1, w1_b, kC * kFF);
  k_cvt<<<dim3(kFF * kC / 256), 256, 0, stream>>>(W2, w2_b, kFF * kC);

  // x (B,C,T) -> h (B,T,C)
  k_transpose<<<dim3(kT / 32, kC / 32, kB), dim3(32, 8), 0, stream>>>(x, h, kC, kT);

  // attention branch
  k_rmsnorm<<<dim3(kM), 256, 0, stream>>>(h, g_attn, a_b);
  dim3 gC(kC / 128, kM / 128);
  k_gemm<0><<<gC, 256, 0, stream>>>(a_b, wq_b, q_b, kC, kC);
  k_gemm<0><<<gC, 256, 0, stream>>>(a_b, wk_b, k_b, kC, kC);
  k_gemm<1><<<gC, 256, 0, stream>>>(a_b, wv_b, vt_b, kC, kC);
  k_flash<<<dim3(kT / 64, kB * kH), 128, 0, stream>>>(q_b, k_b, vt_b, at_b);
  k_gemm<2><<<gC, 256, 0, stream>>>(at_b, wo_b, h, kC, kC);

  // FFN branch
  k_rmsnorm<<<dim3(kM), 256, 0, stream>>>(h, g_ff, f_b);
  dim3 gF(kFF / 128, kM / 128);
  k_gemm<3><<<gF, 256, 0, stream>>>(f_b, w1_b, ff_b, kFF, kC);
  k_gemm<2><<<gC, 256, 0, stream>>>(ff_b, w2_b, h, kC, kFF);

  // h (B,T,C) -> out (B,C,T)
  k_transpose<<<dim3(kC / 32, kT / 32, kB), dim3(32, 8), 0, stream>>>(h, out, kT, kC);
}
